// SSA_24223615549520
// MI455X (gfx1250) — matrix operand order repaired and checked by reading, not compiled, not run on hardware
//
#include <hip/hip_runtime.h>
#include <hip/hip_bf16.h>

typedef __attribute__((ext_vector_type(16))) _Float16 v16h;
typedef __attribute__((ext_vector_type(8)))  _Float16 v8h;
typedef __attribute__((ext_vector_type(8)))  float    v8f;

union AFrag { v16h v; v8h h[2]; };

__device__ __forceinline__ v8f wmma_f16(v16h a, v16h b, v8f c) {
    // D = A(16x32 f16) * B(32x16 f16) + C(16x16 f32)
    return __builtin_amdgcn_wmma_f32_16x16x32_f16(
        false, a, false, b, (short)0, c, false, false);
}

__device__ __forceinline__ float clampf(float v, float lo, float hi) {
    return fminf(fmaxf(v, lo), hi);
}

// ---------- constants ----------
#define CDIM   256
#define HDIM   96
#define HW     9216          // 96*96
#define NPIX   18432         // 2*HW
#define KD     2304          // 256*9
#define OUTB   2359296       // 256*9216

// ---------- elementwise f32 -> f16 ----------
__global__ void k_f32_to_f16(const float* __restrict__ src, _Float16* __restrict__ dst, int n) {
    int i = blockIdx.x * 256 + threadIdx.x;
    if (i < n) dst[i] = (_Float16)src[i];
}

// ---------- x (b,c,hw) -> xT (p=b*HW+hw, c) f16, LDS tile transpose ----------
__global__ void k_transpose_x(const float* __restrict__ x, _Float16* __restrict__ xT) {
    __shared__ float tile[32][33];
    int b = blockIdx.z;
    int hw0 = blockIdx.x * 32, c0 = blockIdx.y * 32;
    int tx = threadIdx.x, ty = threadIdx.y;
    tile[ty][tx] = x[(size_t)(b * CDIM + c0 + ty) * HW + hw0 + tx];
    __syncthreads();
    xT[(size_t)(b * HW + hw0 + ty) * CDIM + c0 + tx] = (_Float16)tile[tx][ty];
}

// ---------- offset conv: 18ch 3x3 pad1, fp32 direct, LDS tiled ----------
__global__ void k_offset_conv(const float* __restrict__ x, const float* __restrict__ wp,
                              const float* __restrict__ bp, float* __restrict__ offs) {
    __shared__ float xs[18][19];
    __shared__ float wl[162];
    int wTile = blockIdx.x, hTile = blockIdx.y, b = blockIdx.z;
    int tid = threadIdx.x;
    int ly = tid >> 4, lx = tid & 15;
    float acc[18];
#pragma unroll
    for (int o = 0; o < 18; ++o) acc[o] = 0.f;
    for (int c = 0; c < CDIM; ++c) {
        __syncthreads();
        if (tid < 162) wl[tid] = wp[(size_t)(tid / 9) * KD + c * 9 + (tid % 9)];
        for (int e = tid; e < 324; e += 256) {
            int yy = e / 18, xx = e % 18;
            int hh = hTile * 16 - 1 + yy, ww = wTile * 16 - 1 + xx;
            xs[yy][xx] = (hh >= 0 && hh < HDIM && ww >= 0 && ww < HDIM)
                         ? x[(size_t)(b * CDIM + c) * HW + hh * HDIM + ww] : 0.f;
        }
        __syncthreads();
#pragma unroll
        for (int i = 0; i < 3; ++i)
#pragma unroll
            for (int j = 0; j < 3; ++j) {
                float xv = xs[ly + i][lx + j];
#pragma unroll
                for (int o = 0; o < 18; ++o) acc[o] += wl[o * 9 + i * 3 + j] * xv;
            }
    }
    int oh = hTile * 16 + ly, ow = wTile * 16 + lx;
#pragma unroll
    for (int o = 0; o < 18; ++o)
        offs[((size_t)(b * 18 + o) * HDIM + oh) * HDIM + ow] = acc[o] + bp[o];
}

// ---------- bilinear meta: per (p,tap) -> 4 gather rows (into xT) + 4 weights ----------
__global__ void k_bilinear_meta(const float* __restrict__ offs,
                                int4* __restrict__ midx, float4* __restrict__ mw) {
    int t = blockIdx.x * 256 + threadIdx.x;
    if (t >= NPIX * 9) return;
    int p = t / 9, tap = t % 9;
    int b = p / HW, hw = p % HW, h = hw / HDIM, w = hw % HDIM;
    float ox = offs[((size_t)(b * 18 + tap) * HDIM + h) * HDIM + w];
    float oy = offs[((size_t)(b * 18 + 9 + tap) * HDIM + h) * HDIM + w];
    float px = ox + (float)(tap / 3 - 1) + (float)(h + 1);
    float py = oy + (float)(tap % 3 - 1) + (float)(w + 1);
    float fx = floorf(px), fy = floorf(py);
    float qltx = clampf(fx, 0.f, 97.f),      qlty = clampf(fy, 0.f, 97.f);
    float qrbx = clampf(fx + 1.f, 0.f, 97.f), qrby = clampf(fy + 1.f, 0.f, 97.f);
    float pxc  = clampf(px, 0.f, 97.f),       pyc  = clampf(py, 0.f, 97.f);
    float glt = (1.f + (qltx - pxc)) * (1.f + (qlty - pyc));
    float grb = (1.f - (qrbx - pxc)) * (1.f - (qrby - pyc));
    float glb = (1.f + (qltx - pxc)) * (1.f - (qrby - pyc));
    float grt = (1.f - (qrbx - pxc)) * (1.f + (qlty - pyc));
    int iltx = (int)qltx, ilty = (int)qlty, irbx = (int)qrbx, irby = (int)qrby;
    auto gidx = [&](int ix, int iy) -> int {
        int xx = ix - 1, yy = iy - 1;
        return (xx >= 0 && xx < HDIM && yy >= 0 && yy < HDIM) ? (b * HW + xx * HDIM + yy) : -1;
    };
    midx[t] = make_int4(gidx(iltx, ilty), gidx(irbx, irby), gidx(iltx, irby), gidx(irbx, ilty));
    mw[t]   = make_float4(glt, grb, glb, grt);
}

// ---------- sample: S[p][c*9+tap] (f16, the dconv B^T operand) ----------
__global__ void k_sample(const _Float16* __restrict__ xT, const int4* __restrict__ midx,
                         const float4* __restrict__ mw, _Float16* __restrict__ S) {
    int p = blockIdx.x * 8 + (threadIdx.x >> 5);       // wave-uniform p
    int c = blockIdx.y * 32 + (threadIdx.x & 31);      // lanes over c: coalesced gathers
#pragma unroll
    for (int tap = 0; tap < 9; ++tap) {
        int4   id = midx[(size_t)p * 9 + tap];
        float4 wv = mw[(size_t)p * 9 + tap];
        float acc = 0.f;
        acc += wv.x * (id.x >= 0 ? (float)xT[(size_t)id.x * CDIM + c] : 0.f);
        acc += wv.y * (id.y >= 0 ? (float)xT[(size_t)id.y * CDIM + c] : 0.f);
        acc += wv.z * (id.z >= 0 ? (float)xT[(size_t)id.z * CDIM + c] : 0.f);
        acc += wv.w * (id.w >= 0 ? (float)xT[(size_t)id.w * CDIM + c] : 0.f);
        S[(size_t)p * KD + c * 9 + tap] = (_Float16)acc;
    }
}

// ---------- generic WMMA GEMM: C[M,N] = A[M,K](row) * BT[N,K](row) ----------
// block = 8 waves; wave computes 16x64; grid (N/64, M/128)
// MODE 0: C[m*ldc+n]  MODE 1: n->(b,hw) split write of d_out
template <typename OutT, int MODE>
__global__ void k_gemm(const _Float16* __restrict__ A, const _Float16* __restrict__ BT,
                       OutT* __restrict__ C, int K, int ldc) {
    const int wave = threadIdx.x >> 5;
    const int lane = threadIdx.x & 31;
    const int l = lane & 15, hi = lane >> 4;
    const int  mTile = (blockIdx.y * 8 + wave) * 16;
    const long nBase = (long)blockIdx.x * 64;
    const _Float16* Arow = A + (size_t)(mTile + l) * K;
    const _Float16* B0   = BT + (size_t)(nBase + l) * K + hi * 16;
    const size_t bstep = (size_t)16 * K;
    v8f acc0 = {}, acc1 = {}, acc2 = {}, acc3 = {};
    for (int k = 0; k < K; k += 32) {
        AFrag a;
        a.h[0] = *(const v8h*)(Arow + k + hi * 16);
        a.h[1] = *(const v8h*)(Arow + k + 8 + hi * 16);
        v16h b0 = *(const v16h*)(B0 + k);
        v16h b1 = *(const v16h*)(B0 + bstep + k);
        v16h b2 = *(const v16h*)(B0 + 2 * bstep + k);
        v16h b3 = *(const v16h*)(B0 + 3 * bstep + k);
        acc0 = wmma_f16(a.v, b0, acc0);
        acc1 = wmma_f16(a.v, b1, acc1);
        acc2 = wmma_f16(a.v, b2, acc2);
        acc3 = wmma_f16(a.v, b3, acc3);
    }
    v8f accs[4] = {acc0, acc1, acc2, acc3};
#pragma unroll
    for (int j = 0; j < 4; ++j) {
#pragma unroll
        for (int r = 0; r < 8; ++r) {
            int  m = mTile + r + hi * 8;
            long n = nBase + j * 16 + l;
            float val = accs[j][r];
            if constexpr (MODE == 0) {
                C[(size_t)m * ldc + n] = (OutT)val;
            } else {
                long bb = n / HW, hw = n - bb * HW;
                ((float*)C)[(size_t)bb * OUTB + (size_t)m * HW + hw] = val;
            }
        }
    }
}

// ---------- per-(channel,batch) L2 norms of kv rows ----------
__global__ void k_rownorm(const float* __restrict__ kv, float* __restrict__ norms) {
    int g = blockIdx.x;               // g = o*2 + b, o in [0,512)
    int o = g >> 1, b = g & 1;
    const float* row = kv + (size_t)o * NPIX + b * HW;
    float s = 0.f;
    for (int i = threadIdx.x; i < HW; i += 256) { float v = row[i]; s += v * v; }
    __shared__ float red[256];
    red[threadIdx.x] = s;
    __syncthreads();
    for (int st = 128; st; st >>= 1) {
        if (threadIdx.x < st) red[threadIdx.x] += red[threadIdx.x + st];
        __syncthreads();
    }
    if (threadIdx.x == 0) norms[g] = fmaxf(sqrtf(red[0]), 1e-12f);
}

// ---------- normalize kv -> k_f16, v_f16 ----------
__global__ void k_normalize(const float* __restrict__ kv, const float* __restrict__ norms,
                            _Float16* __restrict__ kh, _Float16* __restrict__ vh) {
    size_t idx = (size_t)blockIdx.x * 256 + threadIdx.x;
    if (idx >= (size_t)512 * NPIX) return;
    int o = (int)(idx / NPIX), p = (int)(idx % NPIX);
    int g = o * 2 + (p >= HW ? 1 : 0);
    float v = kv[idx] / norms[g];
    if (o < 256) kh[(size_t)o * NPIX + p] = (_Float16)v;
    else         vh[(size_t)(o - 256) * NPIX + p] = (_Float16)v;
}

// ---------- attention logits: attn[b,h,32,32] = q_h * k_h^T  (K=9216 WMMA) ----------
__global__ void k_attn_qk(const _Float16* __restrict__ q, const _Float16* __restrict__ kk,
                          float* __restrict__ attn) {
    int bh = blockIdx.x;              // b*8+h
    int b = bh >> 3, h = bh & 7;
    int wave = threadIdx.x >> 5, lane = threadIdx.x & 31;
    int mi = wave >> 1, ni = wave & 1;
    int l = lane & 15, hi = lane >> 4;
    const _Float16* Arow = q  + (size_t)(h * 32 + mi * 16 + l) * NPIX + b * HW;
    const _Float16* Brow = kk + (size_t)(h * 32 + ni * 16 + l) * NPIX + b * HW + hi * 16;
    v8f acc = {};
    for (int k = 0; k < HW; k += 32) {
        AFrag a;
        a.h[0] = *(const v8h*)(Arow + k + hi * 16);
        a.h[1] = *(const v8h*)(Arow + k + 8 + hi * 16);
        v16h bb = *(const v16h*)(Brow + k);
        acc = wmma_f16(a.v, bb, acc);
    }
#pragma unroll
    for (int r = 0; r < 8; ++r) {
        int m = mi * 16 + r + hi * 8;
        int n = ni * 16 + l;
        attn[((size_t)bh * 32 + m) * 32 + n] = acc[r];
    }
}

// ---------- softmax over last dim (32), with per-head temperature ----------
__global__ void k_softmax(float* __restrict__ attn, const float* __restrict__ temp) {
    int row = blockIdx.x;             // ((b*8+h)*32 + c), 512 rows
    int h = (row >> 5) & 7;
    int lane = threadIdx.x;           // 32 lanes
    float v = attn[(size_t)row * 32 + lane] * temp[h];
    float m = v;
    for (int o = 16; o; o >>= 1) m = fmaxf(m, __shfl_xor(m, o, 32));
    float e = expf(v - m);
    float s = e;
    for (int o = 16; o; o >>= 1) s += __shfl_xor(s, o, 32);
    attn[(size_t)row * 32 + lane] = e / s;
}

// ---------- out^T[p][oc] = sum_d attn[b,h,c,d] * v[h*32+d][p] ----------
__global__ void k_attn_v(const float* __restrict__ attn, const _Float16* __restrict__ v,
                         _Float16* __restrict__ outT) {
    int p = blockIdx.x;
    int oc = threadIdx.x;
    int b = p / HW;
    int h = oc >> 5, c = oc & 31;
    const float* arow = attn + ((size_t)((b * 8 + h) * 32 + c)) * 32;
    float acc = 0.f;
#pragma unroll
    for (int d = 0; d < 32; ++d)
        acc += arow[d] * (float)v[(size_t)(h * 32 + d) * NPIX + p];
    outT[(size_t)p * CDIM + oc] = (_Float16)acc;
}

extern "C" void kernel_launch(void* const* d_in, const int* in_sizes, int n_in,
                              void* d_out, int out_size, void* d_ws, size_t ws_size,
                              hipStream_t stream) {
    const float* x       = (const float*)d_in[0];   // (2,256,96,96)
    const float* w_pconv = (const float*)d_in[1];   // (18,256,3,3)
    const float* b_pconv = (const float*)d_in[2];   // (18,)
    const float* w_dconv = (const float*)d_in[3];   // (256,256,3,3)
    const float* w_qkv   = (const float*)d_in[4];   // (768,256)
    const float* temp    = (const float*)d_in[5];   // (8,1,1)
    const float* w_proj  = (const float*)d_in[6];   // (256,256)
    float* out = (float*)d_out;
    (void)in_sizes; (void)n_in; (void)out_size; (void)ws_size;

    char* base = (char*)d_ws;
    size_t cur = 0;
    auto alloc = [&](size_t bytes) -> void* {
        void* p = base + cur;
        cur = (cur + bytes + 255) & ~(size_t)255;
        return p;
    };

    float*    offs = (float*)   alloc((size_t)2 * 18 * HW * 4);
    int4*     midx = (int4*)    alloc((size_t)NPIX * 9 * 16);
    float4*   mw   = (float4*)  alloc((size_t)NPIX * 9 * 16);
    _Float16* xT   = (_Float16*)alloc((size_t)NPIX * CDIM * 2);
    _Float16* S    = (_Float16*)alloc((size_t)NPIX * KD * 2);     // 85 MB
    _Float16* Wd   = (_Float16*)alloc((size_t)256 * KD * 2);
    _Float16* Wkv  = (_Float16*)alloc((size_t)512 * 256 * 2);
    _Float16* Wpr  = (_Float16*)alloc((size_t)256 * 256 * 2);
    _Float16* qh   = (_Float16*)alloc((size_t)256 * NPIX * 2);
    _Float16* kh   = (_Float16*)alloc((size_t)256 * NPIX * 2);
    _Float16* vh   = (_Float16*)alloc((size_t)256 * NPIX * 2);
    float*    nrm  = (float*)   alloc(1024 * 4);
    float*    attn = (float*)   alloc((size_t)16 * 32 * 32 * 4);
    _Float16* outT = (_Float16*)alloc((size_t)NPIX * CDIM * 2);
    float*    kv   = (float*)S;   // alias: kv GEMM runs after dconv GEMM consumed S

    // weight conversions
    k_f32_to_f16<<<(256 * KD + 255) / 256, 256, 0, stream>>>(w_dconv, Wd, 256 * KD);
    k_f32_to_f16<<<(512 * 256 + 255) / 256, 256, 0, stream>>>(w_qkv + 256 * 256, Wkv, 512 * 256);
    k_f32_to_f16<<<(256 * 256 + 255) / 256, 256, 0, stream>>>(w_proj, Wpr, 256 * 256);

    // x -> xT (f16, [p][c])
    k_transpose_x<<<dim3(HW / 32, CDIM / 32, 2), dim3(32, 32), 0, stream>>>(x, xT);

    // offset conv (fp32)
    k_offset_conv<<<dim3(6, 6, 2), 256, 0, stream>>>(x, w_pconv, b_pconv, offs);

    // bilinear meta + gather into S
    k_bilinear_meta<<<(NPIX * 9 + 255) / 256, 256, 0, stream>>>(offs, midx, mw);
    k_sample<<<dim3(NPIX / 8, CDIM / 32), 256, 0, stream>>>(xT, midx, mw, S);

    // q = Wd (256 x 2304) * S^T   -> qh[o][p] f16
    k_gemm<_Float16, 0><<<dim3(NPIX / 64, 256 / 128), 256, 0, stream>>>(Wd, S, qh, KD, NPIX);

    // kv = Wkv (512 x 256) * xT^T -> kv[o][p] f32 (aliases S; ordered after dconv GEMM)
    k_gemm<float, 0><<<dim3(NPIX / 64, 512 / 128), 256, 0, stream>>>(Wkv, xT, kv, 256, NPIX);

    // row norms + normalize into k/v f16
    k_rownorm<<<1024, 256, 0, stream>>>(kv, nrm);
    k_normalize<<<(int)(((size_t)512 * NPIX + 255) / 256), 256, 0, stream>>>(kv, nrm, kh, vh);

    // attention
    k_attn_qk<<<16, 128, 0, stream>>>(qh, kh, attn);
    k_softmax<<<512, 32, 0, stream>>>(attn, temp);
    k_attn_v<<<NPIX, 256, 0, stream>>>(attn, vh, outT);

    // proj: out = Wpr (256x256) * outT^T, written straight into d_out (B,C,H,W)
    k_gemm<float, 1><<<dim3(NPIX / 64, 256 / 128), 256, 0, stream>>>(Wpr, outT, out, 256, 0);
}
